// EquivariantUpdate_39273180954645
// MI455X (gfx1250) — matrix operand order repaired and checked by reading, not compiled, not run on hardware
//
#include <hip/hip_runtime.h>

#define HID   128
#define EDGES 800000
#define NODES 50000
#define ACT_STRIDE 136   // padded half-stride for LDS activation tile

typedef __attribute__((ext_vector_type(16))) _Float16 v16h;
typedef __attribute__((ext_vector_type(8)))  _Float16 v8h;
typedef __attribute__((ext_vector_type(8)))  float    v8f;

__device__ __forceinline__ float silu_f(float x) { return x / (1.0f + __expf(-x)); }

__device__ __forceinline__ v8h cvt8(const float4 a, const float4 b) {
  v8h r;
  r[0] = (_Float16)a.x; r[1] = (_Float16)a.y; r[2] = (_Float16)a.z; r[3] = (_Float16)a.w;
  r[4] = (_Float16)b.x; r[5] = (_Float16)b.y; r[6] = (_Float16)b.z; r[7] = (_Float16)b.w;
  return r;
}

__device__ __forceinline__ v16h cat16(v8h lo, v8h hi) {
  v16h r;
#pragma unroll
  for (int i = 0; i < 8; ++i) { r[i] = lo[i]; r[8 + i] = hi[i]; }
  return r;
}

// load 16 contiguous halfs (two 16B chunks; 16B-aligned addresses by construction)
__device__ __forceinline__ v16h ldb16h(const _Float16* p) {
  v8h lo = *(const v8h*)p;
  v8h hi = *(const v8h*)(p + 8);
  return cat16(lo, hi);
}

// ---------------------------------------------------------------------------
// Prep: write W1^T (256x128 -> [n][k] f16) and W2^T ([n][k] f16) into ws.
// ---------------------------------------------------------------------------
__global__ void egnn_prep_kernel(const float* __restrict__ W1, const float* __restrict__ W2,
                                 _Float16* __restrict__ W1t, _Float16* __restrict__ W2t) {
  int t = blockIdx.x * blockDim.x + threadIdx.x;
  if (t < 256 * 128) {
    int n = t >> 8, k = t & 255;                 // W1t[n][k] = W1[k][n]
    W1t[n * 256 + k] = (_Float16)W1[k * HID + n];
  } else if (t < 256 * 128 + 128 * 128) {
    int t2 = t - 256 * 128;
    int n = t2 >> 7, k = t2 & 127;               // W2t[n][k] = W2[k][n]
    W2t[n * HID + k] = (_Float16)W2[k * HID + n];
  }
}

// out = coord (atomics accumulate on top)
__global__ void egnn_init_kernel(const float* __restrict__ coord, float* __restrict__ out, int n) {
  int t = blockIdx.x * blockDim.x + threadIdx.x;
  if (t < n) out[t] = coord[t];
}

// ---------------------------------------------------------------------------
// Main edge kernel: one wave32 per 16-edge tile; 8 waves / block.
// ---------------------------------------------------------------------------
__global__ __launch_bounds__(256) void egnn_edge_kernel(
    const float* __restrict__ h, const float* __restrict__ coord_diff,
    const float* __restrict__ edge_attr, const int* __restrict__ edge_index,
    const float* __restrict__ W1, const float* __restrict__ b1,
    const float* __restrict__ b2, const float* __restrict__ W3,
    const _Float16* __restrict__ W1t, const _Float16* __restrict__ W2t,
    float* __restrict__ out) {
  __shared__ __align__(16) _Float16 act[8][16 * ACT_STRIDE];
  __shared__ float philds[8][16];

  const int lane   = threadIdx.x & 31;
  const int wave   = threadIdx.x >> 5;
  const int e0     = (blockIdx.x * 8 + wave) * 16;
  const int mlo    = lane & 15;          // A row / B,C column within 16
  const bool hiH   = lane >= 16;

  // node ids for this lane's A-matrix row (edge e0+mlo)
  const int eA   = e0 + mlo;
  const int rowA = edge_index[eA];
  const int colA = edge_index[EDGES + eA];

  // edge_attr for the 8 C-rows this lane owns: m = r + (hiH ? 8 : 0)
  float ea8[8];
#pragma unroll
  for (int r = 0; r < 8; ++r) ea8[r] = edge_attr[e0 + r + (hiH ? 8 : 0)];

  // ---- layer 1 C init: b1[n] + edge_attr[m] * W1[256][n] (rank-1 fold of 257th input)
  v8f acc[8];
#pragma unroll
  for (int nb = 0; nb < 8; ++nb) {
    const int   n  = nb * 16 + mlo;
    const float bv = b1[n];
    const float wl = W1[256 * HID + n];
#pragma unroll
    for (int r = 0; r < 8; ++r) acc[nb][r] = bv + ea8[r] * wl;
  }

  // ---- layer 1: [16x256] x [256x128], K in 8 steps of 32 (kk<4 -> h[row], else h[col])
#pragma unroll
  for (int kk = 0; kk < 8; ++kk) {
    const int    node = (kk < 4) ? rowA : colA;
    const float* p    = h + (long)node * HID + ((kk & 3) * 32) + (hiH ? 8 : 0);
    float4 f0 = *(const float4*)(p);
    float4 f1 = *(const float4*)(p + 4);
    float4 f2 = *(const float4*)(p + 16);
    float4 f3 = *(const float4*)(p + 20);
    v16h afrag = cat16(cvt8(f0, f1), cvt8(f2, f3));
#pragma unroll
    for (int nb = 0; nb < 8; ++nb) {
      const _Float16* bp = W1t + (nb * 16 + mlo) * 256 + kk * 32 + (hiH ? 8 : 0);
      v16h bfrag = cat16(*(const v8h*)bp, *(const v8h*)(bp + 16));
      acc[nb] = __builtin_amdgcn_wmma_f32_16x16x32_f16(
          false, afrag, false, bfrag, (short)0, acc[nb], false, false);
    }
  }

  // ---- silu + stage x1 to LDS as f16 row-major [m][n]
#pragma unroll
  for (int nb = 0; nb < 8; ++nb) {
    const int n = nb * 16 + mlo;
#pragma unroll
    for (int r = 0; r < 8; ++r) {
      const int m = r + (hiH ? 8 : 0);
      act[wave][m * ACT_STRIDE + n] = (_Float16)silu_f(acc[nb][r]);
    }
  }
  __syncthreads();   // uniform across waves; orders LDS store->load

  // ---- layer 2: [16x128] x [128x128]
  v8f acc2[8];
#pragma unroll
  for (int nb = 0; nb < 8; ++nb) {
    const float bv = b2[nb * 16 + mlo];
#pragma unroll
    for (int r = 0; r < 8; ++r) acc2[nb][r] = bv;
  }
#pragma unroll
  for (int kk = 0; kk < 4; ++kk) {
    const _Float16* ap = &act[wave][mlo * ACT_STRIDE + kk * 32 + (hiH ? 8 : 0)];
    v16h afrag = cat16(*(const v8h*)ap, *(const v8h*)(ap + 16));
#pragma unroll
    for (int nb = 0; nb < 8; ++nb) {
      const _Float16* bp = W2t + (nb * 16 + mlo) * HID + kk * 32 + (hiH ? 8 : 0);
      v16h bfrag = cat16(*(const v8h*)bp, *(const v8h*)(bp + 16));
      acc2[nb] = __builtin_amdgcn_wmma_f32_16x16x32_f16(
          false, afrag, false, bfrag, (short)0, acc2[nb], false, false);
    }
  }

  // ---- layer 3: phi[m] = sum_n silu(x2[m][n]) * W3[n]; reduce across the 16-lane half
  float part[8];
#pragma unroll
  for (int r = 0; r < 8; ++r) part[r] = 0.0f;
#pragma unroll
  for (int nb = 0; nb < 8; ++nb) {
    const float w3 = W3[nb * 16 + mlo];
#pragma unroll
    for (int r = 0; r < 8; ++r) part[r] += silu_f(acc2[nb][r]) * w3;
  }
#pragma unroll
  for (int mask = 1; mask < 16; mask <<= 1) {
#pragma unroll
    for (int r = 0; r < 8; ++r) part[r] += __shfl_xor(part[r], mask, 32);
  }
  if (mlo == 0) {
#pragma unroll
    for (int r = 0; r < 8; ++r) philds[wave][(hiH ? 8 : 0) + r] = part[r];
  }
  __syncthreads();

  // ---- epilogue: trans = coord_diff * phi / 100, atomic segment-sum over row
  if (lane < 16) {
    const int   e  = e0 + lane;
    const float t  = philds[wave][lane] * 0.01f;   // / NORM_FACTOR
    const int   rw = edge_index[e];
    const float c0 = coord_diff[(long)e * 3 + 0];
    const float c1 = coord_diff[(long)e * 3 + 1];
    const float c2 = coord_diff[(long)e * 3 + 2];
    atomicAdd(&out[(long)rw * 3 + 0], c0 * t);
    atomicAdd(&out[(long)rw * 3 + 1], c1 * t);
    atomicAdd(&out[(long)rw * 3 + 2], c2 * t);
  }
}

extern "C" void kernel_launch(void* const* d_in, const int* in_sizes, int n_in,
                              void* d_out, int out_size, void* d_ws, size_t ws_size,
                              hipStream_t stream) {
  (void)in_sizes; (void)n_in; (void)out_size; (void)ws_size;
  const float* h          = (const float*)d_in[0];
  const float* coord      = (const float*)d_in[1];
  const float* coord_diff = (const float*)d_in[2];
  /* d_in[3] coord_cross unused (reflection_equiv=True) */
  const float* edge_attr  = (const float*)d_in[4];
  const int*   edge_index = (const int*)d_in[5];
  const float* W1         = (const float*)d_in[6];
  const float* b1         = (const float*)d_in[7];
  const float* W2         = (const float*)d_in[8];
  const float* b2         = (const float*)d_in[9];
  const float* W3         = (const float*)d_in[10];
  float*       out        = (float*)d_out;

  _Float16* W1t = (_Float16*)d_ws;                              // 256*128 f16 = 64 KB
  _Float16* W2t = (_Float16*)((char*)d_ws + 256 * 128 * 2);     // 128*128 f16 = 32 KB

  const int prepN = 256 * 128 + 128 * 128;
  egnn_prep_kernel<<<(prepN + 255) / 256, 256, 0, stream>>>(W1, W2, W1t, W2t);
  egnn_init_kernel<<<(NODES * 3 + 255) / 256, 256, 0, stream>>>(coord, out, NODES * 3);
  egnn_edge_kernel<<<EDGES / 128, 256, 0, stream>>>(h, coord_diff, edge_attr, edge_index,
                                                    W1, b1, b2, W3, W1t, W2t, out);
}
